// TextEncoder_45938970198175
// MI455X (gfx1250) — compile-verified, matrix operand order checked by reading, not run on hardware
//
#include <hip/hip_runtime.h>

// ---------------------------------------------------------------------------
// Problem constants (from reference)
// ---------------------------------------------------------------------------
#define LNUM   2
#define HNUM   16
#define DHEAD  128
#define DMODEL 2048
#define FDIM   8192
#define NBLK   64          // NF4 absmax block
#define BATCH  16
#define SEQ    77
#define RLORA  16
#define MTOK   (BATCH * SEQ)   // 1232
#define LORA_SCALE 2.0f

#if defined(__has_builtin)
#if __has_builtin(__builtin_amdgcn_global_load_async_to_lds_b128)
#define HAVE_ASYNC_LDS 1
#endif
#endif
#ifndef HAVE_ASYNC_LDS
#define HAVE_ASYNC_LDS 0
#endif

#define AS1 __attribute__((address_space(1)))
#define AS3 __attribute__((address_space(3)))

typedef __attribute__((ext_vector_type(16))) _Float16 v16h;
typedef __attribute__((ext_vector_type(8)))  _Float16 v8h;
typedef __attribute__((ext_vector_type(8)))  float    v8f;
typedef int v4ig __attribute__((vector_size(16)));   // matches builtin proto

union U16 { v16h v; v8h h[2]; };

__device__ __constant__ float kNF4[16] = {
    -1.0f, -0.6961928009986877f, -0.5250730514526367f, -0.39491748809814453f,
    -0.28444138169288635f, -0.18477343022823334f, -0.09105003625154495f, 0.0f,
    0.07958029955625534f, 0.16093020141124725f, 0.24611230194568634f,
    0.33791524171829224f, 0.44070982933044434f, 0.5626170039176941f,
    0.723855197429657f, 1.0f};

// ---------------------------------------------------------------------------
// Embedding gather: h[m, :] = embed[ids[m], :]
// ---------------------------------------------------------------------------
__global__ void embed_kernel(float* __restrict__ h, const float* __restrict__ embed,
                             const int* __restrict__ ids) {
    int m = blockIdx.x;
    int id = ids[m];
    const float* src = embed + (size_t)id * DMODEL;
    float* dst = h + (size_t)m * DMODEL;
    for (int d = threadIdx.x; d < DMODEL; d += blockDim.x) dst[d] = src[d];
}

// ---------------------------------------------------------------------------
// RMSNorm one row per block
// ---------------------------------------------------------------------------
__global__ void rmsnorm_kernel(float* __restrict__ out, const float* __restrict__ in,
                               const float* __restrict__ w, int ncols) {
    __shared__ float red[256];
    int m = blockIdx.x, t = threadIdx.x;
    const float* row = in + (size_t)m * ncols;
    float ss = 0.f;
    for (int d = t; d < ncols; d += 256) { float v = row[d]; ss += v * v; }
    red[t] = ss; __syncthreads();
    for (int off = 128; off > 0; off >>= 1) {
        if (t < off) red[t] += red[t + off];
        __syncthreads();
    }
    float scale = rsqrtf(red[0] / (float)ncols + 1e-6f);
    float* orow = out + (size_t)m * ncols;
    for (int d = t; d < ncols; d += 256) orow[d] = row[d] * scale * w[d];
}

// ---------------------------------------------------------------------------
// LoRA first stage: T[m, r] = sum_k X[m,k] * A[r,k]   (R = 16)
// ---------------------------------------------------------------------------
__global__ void lora_t_kernel(float* __restrict__ T, const float* __restrict__ X,
                              const float* __restrict__ A, int Kdim) {
    __shared__ float red[128];
    int m = blockIdx.x, r = blockIdx.y, t = threadIdx.x;
    const float* xr = X + (size_t)m * Kdim;
    const float* ar = A + (size_t)r * Kdim;
    float s = 0.f;
    for (int k = t; k < Kdim; k += 128) s += xr[k] * ar[k];
    red[t] = s; __syncthreads();
    for (int off = 64; off > 0; off >>= 1) {
        if (t < off) red[t] += red[t + off];
        __syncthreads();
    }
    if (t == 0) T[(size_t)m * RLORA + r] = red[0];
}

// ---------------------------------------------------------------------------
// Fused NF4-dequant GEMM + LoRA epilogue (+ optional residual).
//   Y[m,n] = sum_k X[m,k]*dequant(W)[n,k] + 2*sum_r T[m,r]*Bl[n,r] (+resid)
//
// Block tile 128(M) x 128(N), K-step 64 (one NF4 absmax block).
// 8 waves in 4(M) x 2(N); each wave owns 32x64 = 2x4 WMMA 16x16 subtiles.
// Pipeline: NF4 code tiles are DMA'd global->LDS with
// GLOBAL_LOAD_ASYNC_TO_LDS_B128 (ASYNCcnt) while the previous tile's WMMAs
// run; X tile is register-staged (needs f32->f16 convert anyway).
// WMMA 16-bit operand layout (ISA 7.12.2): lane<16 holds K {0-7,16-23},
// lane>=16 holds K {8-15,24-31}; B supplied column-major as N x K.
// ---------------------------------------------------------------------------
__global__ __launch_bounds__(256) void qlora_gemm_kernel(
    float* __restrict__ Y, const float* __restrict__ Xg,
    const int* __restrict__ codes, const float* __restrict__ absmax,
    const float* __restrict__ Tl, const float* __restrict__ Bl,
    const float* __restrict__ resid, int Mrows, int Ncols, int Kdim) {

    __shared__ __align__(16) int      rawC[128 * 64];   // 32 KB raw codes
    __shared__ __align__(16) _Float16 sX[128 * 64];     // 16 KB f16 X tile
    __shared__ __align__(16) _Float16 sW[128 * 64];     // 16 KB f16 W tile

    const int t    = threadIdx.x;
    const int lane = t & 31;
    const int wave = t >> 5;
    const int wm   = wave >> 1;            // 0..3  (M quadrant, 32 rows)
    const int wn   = wave & 1;             // 0..1  (N half, 64 cols)
    const int mb0  = blockIdx.x * 128;
    const int nb0  = blockIdx.y * 128;

    // per-thread vec4 slots covering a 128x64 tile (8 groups of 4)
    int rloc[8], kloc[8];
    #pragma unroll
    for (int i = 0; i < 8; ++i) {
        int flat = t + i * 256;
        rloc[i] = flat >> 4;               // tile row 0..127
        kloc[i] = (flat & 15) << 2;        // tile k 0..60 step 4
    }

    const v8f vz = {0.f, 0.f, 0.f, 0.f, 0.f, 0.f, 0.f, 0.f};
    v8f acc[2][4];
    #pragma unroll
    for (int mi = 0; mi < 2; ++mi)
        #pragma unroll
        for (int ni = 0; ni < 4; ++ni) acc[mi][ni] = vz;

    float4 xv[8];                          // staged X tile
#if !HAVE_ASYNC_LDS
    int4 cv[8];                            // staged code tile (fallback)
#endif

    // ---- issue global loads for tile at kb (codes async -> LDS, X -> regs)
    auto issue_loads = [&](int kb) {
        #pragma unroll
        for (int i = 0; i < 8; ++i) {
            int gm  = mb0 + rloc[i];
            int gmc = (gm < Mrows) ? gm : (Mrows - 1);   // clamp: no exec divergence
            xv[i] = *(const float4*)(Xg + (size_t)gmc * Kdim + kb + kloc[i]);
        }
        #pragma unroll
        for (int i = 0; i < 8; ++i) {
            const int* src = codes + (size_t)(nb0 + rloc[i]) * Kdim + kb + kloc[i];
#if HAVE_ASYNC_LDS
            __builtin_amdgcn_global_load_async_to_lds_b128(
                (AS1 v4ig*)src, (AS3 v4ig*)&rawC[rloc[i] * 64 + kloc[i]], 0, 0);
#else
            cv[i] = *(const int4*)src;
#endif
        }
    };

    // ---- dequant / convert staged tile into sX / sW
    auto commit_tile = [&](int kb) {
        #pragma unroll
        for (int i = 0; i < 8; ++i) {
            bool ok = (mb0 + rloc[i]) < Mrows;
            float4 v = xv[i];
            _Float16* d = &sX[rloc[i] * 64 + kloc[i]];
            d[0] = (_Float16)(ok ? v.x : 0.f);
            d[1] = (_Float16)(ok ? v.y : 0.f);
            d[2] = (_Float16)(ok ? v.z : 0.f);
            d[3] = (_Float16)(ok ? v.w : 0.f);
        }
        #pragma unroll
        for (int i = 0; i < 8; ++i) {
#if HAVE_ASYNC_LDS
            int4 cd = *(const int4*)&rawC[rloc[i] * 64 + kloc[i]];
#else
            int4 cd = cv[i];
#endif
            float sc = absmax[(size_t)(nb0 + rloc[i]) * (Kdim >> 6) + (kb >> 6)];
            _Float16* d = &sW[rloc[i] * 64 + kloc[i]];
            d[0] = (_Float16)(kNF4[cd.x & 15] * sc);
            d[1] = (_Float16)(kNF4[cd.y & 15] * sc);
            d[2] = (_Float16)(kNF4[cd.z & 15] * sc);
            d[3] = (_Float16)(kNF4[cd.w & 15] * sc);
        }
    };

    issue_loads(0);

    for (int kb = 0; kb < Kdim; kb += 64) {
#if HAVE_ASYNC_LDS
#if defined(__has_builtin) && __has_builtin(__builtin_amdgcn_s_wait_asynccnt)
        __builtin_amdgcn_s_wait_asynccnt(0);
#else
        asm volatile("s_wait_asynccnt 0x0" ::: "memory");
#endif
#endif
        __syncthreads();                   // raw tile complete; prev compute done
        commit_tile(kb);
        __syncthreads();                   // sX / sW published
        if (kb + 64 < Kdim) {
            issue_loads(kb + 64);          // overlap next DMA with WMMA stage
            __builtin_prefetch(Xg + (size_t)(mb0 + rloc[0]) * Kdim + kb + 64, 0, 0);
        }

        #pragma unroll
        for (int kc = 0; kc < 64; kc += 32) {
            const int koff = kc + ((lane & 16) ? 8 : 0);
            const int mrow = wm * 32 + (lane & 15);
            const int ncol = wn * 64 + (lane & 15);
            U16 a0, a1, b0, b1, b2, b3;
            a0.h[0] = *(const v8h*)&sX[(mrow)      * 64 + koff];
            a0.h[1] = *(const v8h*)&sX[(mrow)      * 64 + koff + 16];
            a1.h[0] = *(const v8h*)&sX[(mrow + 16) * 64 + koff];
            a1.h[1] = *(const v8h*)&sX[(mrow + 16) * 64 + koff + 16];
            b0.h[0] = *(const v8h*)&sW[(ncol)      * 64 + koff];
            b0.h[1] = *(const v8h*)&sW[(ncol)      * 64 + koff + 16];
            b1.h[0] = *(const v8h*)&sW[(ncol + 16) * 64 + koff];
            b1.h[1] = *(const v8h*)&sW[(ncol + 16) * 64 + koff + 16];
            b2.h[0] = *(const v8h*)&sW[(ncol + 32) * 64 + koff];
            b2.h[1] = *(const v8h*)&sW[(ncol + 32) * 64 + koff + 16];
            b3.h[0] = *(const v8h*)&sW[(ncol + 48) * 64 + koff];
            b3.h[1] = *(const v8h*)&sW[(ncol + 48) * 64 + koff + 16];
            acc[0][0] = __builtin_amdgcn_wmma_f32_16x16x32_f16(false, a0.v, false, b0.v,
                                                               (short)0, acc[0][0], false, false);
            acc[0][1] = __builtin_amdgcn_wmma_f32_16x16x32_f16(false, a0.v, false, b1.v,
                                                               (short)0, acc[0][1], false, false);
            acc[0][2] = __builtin_amdgcn_wmma_f32_16x16x32_f16(false, a0.v, false, b2.v,
                                                               (short)0, acc[0][2], false, false);
            acc[0][3] = __builtin_amdgcn_wmma_f32_16x16x32_f16(false, a0.v, false, b3.v,
                                                               (short)0, acc[0][3], false, false);
            acc[1][0] = __builtin_amdgcn_wmma_f32_16x16x32_f16(false, a1.v, false, b0.v,
                                                               (short)0, acc[1][0], false, false);
            acc[1][1] = __builtin_amdgcn_wmma_f32_16x16x32_f16(false, a1.v, false, b1.v,
                                                               (short)0, acc[1][1], false, false);
            acc[1][2] = __builtin_amdgcn_wmma_f32_16x16x32_f16(false, a1.v, false, b2.v,
                                                               (short)0, acc[1][2], false, false);
            acc[1][3] = __builtin_amdgcn_wmma_f32_16x16x32_f16(false, a1.v, false, b3.v,
                                                               (short)0, acc[1][3], false, false);
        }
    }

    // ---- epilogue: C/D layout VGPR r -> M = r (+8 upper lanes), N = lane&15
    #pragma unroll
    for (int mi = 0; mi < 2; ++mi) {
        #pragma unroll
        for (int ni = 0; ni < 4; ++ni) {
            int gn = nb0 + wn * 64 + ni * 16 + (lane & 15);
            float bcol[16];
            #pragma unroll
            for (int j = 0; j < 16; ++j) bcol[j] = Bl[(size_t)gn * RLORA + j];
            #pragma unroll
            for (int r = 0; r < 8; ++r) {
                int gm = mb0 + wm * 32 + mi * 16 + ((lane & 16) ? (r + 8) : r);
                if (gm < Mrows) {
                    float lora = 0.f;
                    #pragma unroll
                    for (int j = 0; j < 16; ++j)
                        lora += Tl[(size_t)gm * RLORA + j] * bcol[j];
                    float val = acc[mi][ni][r] + LORA_SCALE * lora;
                    size_t idx = (size_t)gm * Ncols + gn;
                    if (resid) val += resid[idx];
                    Y[idx] = val;
                }
            }
        }
    }
}

// ---------------------------------------------------------------------------
// RoPE + causal attention. One block per (b, head), 128 threads.
// K rows (rope-applied) cached in LDS; V read from global (L2 resident).
// ---------------------------------------------------------------------------
__global__ void attn_kernel(float* __restrict__ aout, const float* __restrict__ Qg,
                            const float* __restrict__ Kg, const float* __restrict__ Vg,
                            const int* __restrict__ amask) {
    __shared__ float kS[SEQ * DHEAD];   // ~39.4 KB
    __shared__ float qrow[DHEAD];
    __shared__ float pS[128];
    __shared__ float sR[128];

    const int bh = blockIdx.x;
    const int b = bh >> 4, head = bh & 15;
    const int t = threadIdx.x;
    const float kInv = 0.14391156f;                 // ln(10000)/64
    const float kScale = 0.08838834764831845f;      // 1/sqrt(128)

    for (int idx = t; idx < SEQ * DHEAD; idx += 128) {
        int sp = idx >> 7, d = idx & 127;
        size_t base = ((size_t)(b * SEQ + sp)) * DMODEL + head * DHEAD;
        float x1 = Kg[base + d];
        int j = (d < 64) ? d : d - 64;
        float partner = Kg[base + ((d < 64) ? d + 64 : d - 64)];
        float ang = (float)sp * __expf(-(float)j * kInv);
        float rot = (d < 64) ? -partner : partner;
        kS[idx] = x1 * __cosf(ang) + rot * __sinf(ang);
    }
    __syncthreads();

    for (int qi = 0; qi < SEQ; ++qi) {
        if (t < DHEAD) {
            size_t base = ((size_t)(b * SEQ + qi)) * DMODEL + head * DHEAD;
            int d = t;
            float x1 = Qg[base + d];
            int j = (d < 64) ? d : d - 64;
            float partner = Qg[base + ((d < 64) ? d + 64 : d - 64)];
            float ang = (float)qi * __expf(-(float)j * kInv);
            float rot = (d < 64) ? -partner : partner;
            qrow[d] = x1 * __cosf(ang) + rot * __sinf(ang);
        }
        __syncthreads();

        float sc = -1e9f;
        if (t < SEQ && t <= qi && amask[b * SEQ + t] != 0) {
            float dot = 0.f;
            #pragma unroll 4
            for (int d = 0; d < DHEAD; ++d) dot += qrow[d] * kS[t * DHEAD + d];
            sc = dot * kScale;
        }
        pS[t] = sc; sR[t] = sc;
        __syncthreads();
        for (int off = 64; off > 0; off >>= 1) {
            if (t < off) sR[t] = fmaxf(sR[t], sR[t + off]);
            __syncthreads();
        }
        float mx = sR[0];
        __syncthreads();
        float ex = (t < SEQ) ? __expf(pS[t] - mx) : 0.f;
        pS[t] = ex; sR[t] = ex;
        __syncthreads();
        for (int off = 64; off > 0; off >>= 1) {
            if (t < off) sR[t] += sR[t + off];
            __syncthreads();
        }
        float inv = 1.f / sR[0];
        __syncthreads();
        if (t < DHEAD) {
            float o = 0.f;
            for (int j2 = 0; j2 < SEQ; ++j2) {
                size_t vb = ((size_t)(b * SEQ + j2)) * DMODEL + head * DHEAD;
                o += pS[j2] * Vg[vb + t];
            }
            aout[((size_t)(b * SEQ + qi)) * DMODEL + head * DHEAD + t] = o * inv;
        }
        __syncthreads();
    }
}

// ---------------------------------------------------------------------------
// g = silu(g) * up
// ---------------------------------------------------------------------------
__global__ void silu_mul_kernel(float* __restrict__ g, const float* __restrict__ up, int n) {
    int i = blockIdx.x * 256 + threadIdx.x;
    if (i < n) {
        float gv = g[i];
        g[i] = (gv / (1.f + __expf(-gv))) * up[i];
    }
}

// ---------------------------------------------------------------------------
// Final: rmsnorm(h, ln_final), pick last non-pad token per batch
// ---------------------------------------------------------------------------
__global__ void final_kernel(float* __restrict__ out, const float* __restrict__ h,
                             const float* __restrict__ lnf, const int* __restrict__ amask) {
    __shared__ float red[256];
    __shared__ int sLen;
    int b = blockIdx.x, t = threadIdx.x;
    if (t == 0) {
        int len = 0;
        for (int j = 0; j < SEQ; ++j) len += amask[b * SEQ + j];
        sLen = len - 1;
    }
    __syncthreads();
    const float* row = h + ((size_t)(b * SEQ + sLen)) * DMODEL;
    float ss = 0.f;
    for (int d = t; d < DMODEL; d += 256) { float v = row[d]; ss += v * v; }
    red[t] = ss; __syncthreads();
    for (int off = 128; off > 0; off >>= 1) {
        if (t < off) red[t] += red[t + off];
        __syncthreads();
    }
    float scale = rsqrtf(red[0] / (float)DMODEL + 1e-6f);
    for (int d = t; d < DMODEL; d += 256)
        out[(size_t)b * DMODEL + d] = row[d] * scale * lnf[d];
}

// ---------------------------------------------------------------------------
// Host orchestration
// ---------------------------------------------------------------------------
extern "C" void kernel_launch(void* const* d_in, const int* in_sizes, int n_in,
                              void* d_out, int out_size, void* d_ws, size_t ws_size,
                              hipStream_t stream) {
    const int*   input_ids   = (const int*)d_in[0];
    const int*   amask       = (const int*)d_in[1];
    const float* embed       = (const float*)d_in[2];
    const float* ln_attn     = (const float*)d_in[3];
    const float* ln_mlp      = (const float*)d_in[4];
    const float* ln_final    = (const float*)d_in[5];
    const int*   qkv_codes   = (const int*)d_in[6];
    const float* qkv_absmax  = (const float*)d_in[7];
    const float* qkv_A       = (const float*)d_in[8];
    const float* qkv_B       = (const float*)d_in[9];
    const int*   o_codes     = (const int*)d_in[10];
    const float* o_absmax    = (const float*)d_in[11];
    const float* o_A         = (const float*)d_in[12];
    const float* o_B         = (const float*)d_in[13];
    const int*   gu_codes    = (const int*)d_in[14];
    const float* gu_absmax   = (const float*)d_in[15];
    const float* gu_A        = (const float*)d_in[16];
    const float* gu_B        = (const float*)d_in[17];
    const int*   down_codes  = (const int*)d_in[18];
    const float* down_absmax = (const float*)d_in[19];
    const float* down_A      = (const float*)d_in[20];
    const float* down_B      = (const float*)d_in[21];

    const size_t MD = (size_t)MTOK * DMODEL;
    const size_t MF = (size_t)MTOK * FDIM;
    float* w    = (float*)d_ws;
    float* h    = w;            // M x D
    float* xn   = h   + MD;     // M x D
    float* qb   = xn  + MD;     // M x D
    float* kbuf = qb  + MD;     // M x D
    float* vb   = kbuf + MD;    // M x D
    float* ao   = vb  + MD;     // M x D (attention out)
    float* gbuf = ao  + MD;     // M x F
    float* ubuf = gbuf + MF;    // M x F
    float* Tb   = ubuf + MF;    // M x R

    const int gmx = (MTOK + 127) / 128;

    embed_kernel<<<MTOK, 256, 0, stream>>>(h, embed, input_ids);

    for (int l = 0; l < LNUM; ++l) {
        // ---- attention block
        rmsnorm_kernel<<<MTOK, 256, 0, stream>>>(xn, h, ln_attn + (size_t)l * DMODEL, DMODEL);
        float* qkvOut[3] = {qb, kbuf, vb};
        for (int i = 0; i < 3; ++i) {
            size_t off = (size_t)l * 3 + i;
            lora_t_kernel<<<dim3(MTOK, RLORA), 128, 0, stream>>>(
                Tb, xn, qkv_A + off * RLORA * DMODEL, DMODEL);
            qlora_gemm_kernel<<<dim3(gmx, DMODEL / 128), 256, 0, stream>>>(
                qkvOut[i], xn,
                qkv_codes + off * (size_t)DMODEL * DMODEL,
                qkv_absmax + off * (size_t)DMODEL * (DMODEL / NBLK),
                Tb, qkv_B + off * (size_t)DMODEL * RLORA,
                nullptr, MTOK, DMODEL, DMODEL);
        }
        attn_kernel<<<BATCH * HNUM, 128, 0, stream>>>(ao, qb, kbuf, vb, amask);
        lora_t_kernel<<<dim3(MTOK, RLORA), 128, 0, stream>>>(
            Tb, ao, o_A + (size_t)l * RLORA * DMODEL, DMODEL);
        qlora_gemm_kernel<<<dim3(gmx, DMODEL / 128), 256, 0, stream>>>(
            h, ao,
            o_codes + (size_t)l * DMODEL * DMODEL,
            o_absmax + (size_t)l * DMODEL * (DMODEL / NBLK),
            Tb, o_B + (size_t)l * DMODEL * RLORA,
            h, MTOK, DMODEL, DMODEL);

        // ---- MLP block
        rmsnorm_kernel<<<MTOK, 256, 0, stream>>>(xn, h, ln_mlp + (size_t)l * DMODEL, DMODEL);
        for (int i = 0; i < 2; ++i) {       // 0 = gate, 1 = up
            size_t off = (size_t)l * 2 + i;
            lora_t_kernel<<<dim3(MTOK, RLORA), 128, 0, stream>>>(
                Tb, xn, gu_A + off * RLORA * DMODEL, DMODEL);
            qlora_gemm_kernel<<<dim3(gmx, FDIM / 128), 256, 0, stream>>>(
                (i == 0 ? gbuf : ubuf), xn,
                gu_codes + off * (size_t)FDIM * DMODEL,
                gu_absmax + off * (size_t)FDIM * (DMODEL / NBLK),
                Tb, gu_B + off * (size_t)FDIM * RLORA,
                nullptr, MTOK, FDIM, DMODEL);
        }
        silu_mul_kernel<<<(int)((MF + 255) / 256), 256, 0, stream>>>(gbuf, ubuf, (int)MF);
        lora_t_kernel<<<dim3(MTOK, RLORA), 128, 0, stream>>>(
            Tb, gbuf, down_A + (size_t)l * RLORA * FDIM, FDIM);
        qlora_gemm_kernel<<<dim3(gmx, DMODEL / 128), 256, 0, stream>>>(
            h, gbuf,
            down_codes + (size_t)l * DMODEL * FDIM,
            down_absmax + (size_t)l * DMODEL * (FDIM / NBLK),
            Tb, down_B + (size_t)l * DMODEL * RLORA,
            h, MTOK, DMODEL, FDIM);
    }

    final_kernel<<<BATCH, 256, 0, stream>>>((float*)d_out, h, ln_final, amask);
    (void)in_sizes; (void)n_in; (void)out_size; (void)ws_size;
}
